// Mamba2Block_44959717654936
// MI455X (gfx1250) — matrix operand order repaired and checked by reading, not compiled, not run on hardware
//
#include <hip/hip_runtime.h>
#include <hip/hip_bf16.h>

// ---------------------------------------------------------------------------
// Mamba2 block forward for MI455X (gfx1250, wave32, WMMA).
//
// Pipeline:
//   1. convert x -> bf16 (row-major, A-side of WMMA)
//   2. convert+transpose weights -> bf16 N x K ("BT", B-side of WMMA)
//      Win padded from N=648 to N=768 with zero columns so all GEMM dims
//      are multiples of the 128x128 block tile (no bounds checks anywhere).
//   3. GEMM1: zs  = silu(x @ Wz)            (bf16 out, fused SiLU epilogue)
//   4. GEMM2: ssm = x @ Win                  (f32 out, ld = 768)
//   5. scan : sequential SSD recurrence, 16 blocks (one per (batch,head)),
//             state h[64][64] in registers (16 f32 per thread), y -> f32,
//             h_last -> tail of d_out.
//   6. GEMM3: gb  = bf16( (zs @ Wzg) * y )   (fused elementwise-y epilogue)
//   7. GEMM4: out = gb @ Wout                (f32 into d_out)
// ---------------------------------------------------------------------------

typedef __bf16 bf16_t;
typedef __bf16 v16bf __attribute__((ext_vector_type(16)));
typedef __bf16 v8bf  __attribute__((ext_vector_type(8)));
typedef float  v8f   __attribute__((ext_vector_type(8)));

#define BATCH   2
#define SEQ     4096
#define MTOT    (BATCH * SEQ)      // 8192 rows
#define DMODEL  1024
#define DZ      2048               // D_MODEL * EXPAND
#define DINNER  512                // N_HEADS * HEAD_DIM
#define NHEADS  8
#define DSTATE  64
#define HEADDIM 64
#define NSSM    648                // 2*DSTATE + NHEADS + DINNER
#define LDSSM   768                // padded ssm width (multiple of 128)

// ---------------------------------------------------------------------------
// Elementwise f32 -> bf16 conversion (row-major copy)
// ---------------------------------------------------------------------------
__global__ __launch_bounds__(256) void cvt_bf16_kernel(
    const float* __restrict__ in, bf16_t* __restrict__ out, size_t n) {
  size_t i = (size_t)blockIdx.x * blockDim.x + threadIdx.x;
  size_t stride = (size_t)gridDim.x * blockDim.x;
  for (; i < n; i += stride) out[i] = (bf16_t)in[i];
}

// W is K x N (f32, row-major). WT is Npad x K (bf16). Rows n >= N are zero.
__global__ __launch_bounds__(256) void transpose_cvt_kernel(
    const float* __restrict__ W, bf16_t* __restrict__ WT,
    int K, int N, int Npad) {
  size_t total = (size_t)Npad * K;
  size_t i = (size_t)blockIdx.x * blockDim.x + threadIdx.x;
  size_t stride = (size_t)gridDim.x * blockDim.x;
  for (; i < total; i += stride) {
    int n = (int)(i / K);
    int k = (int)(i % K);
    WT[i] = (n < N) ? (bf16_t)W[(size_t)k * N + n] : (bf16_t)0.0f;
  }
}

// ---------------------------------------------------------------------------
// bf16 WMMA GEMM: C[M,N] = A[M,K] * BT[N,K]^T with f32 accumulation.
// Block = 256 threads = 8 waves, block tile 128(M) x 128(N),
// each wave: 32(M) x 64(N) = 2x4 v_wmma_f32_16x16x32_bf16 accumulators.
// All of M, N, K are multiples of the tile sizes (by construction).
//
// Epilogue modes:
//   EPI==0 : outF[row*ldOut+col] = acc
//   EPI==1 : outB[row*ldOut+col] = bf16(silu(acc))        (fast rcp path)
//   EPI==2 : outB[row*ldOut+col] = bf16(acc * Y[row*ldOut+col])
// ---------------------------------------------------------------------------
__device__ inline v16bf cat8(v8bf lo, v8bf hi) {
  v16bf r;
#pragma unroll
  for (int i = 0; i < 8; ++i) { r[i] = lo[i]; r[i + 8] = hi[i]; }
  return r;
}

template <int EPI>
__global__ __launch_bounds__(256) void gemm_bf16_kernel(
    const bf16_t* __restrict__ A,   // M x K row-major
    const bf16_t* __restrict__ BT,  // N x K row-major (transposed weight)
    int K,
    float*  __restrict__ outF,
    bf16_t* __restrict__ outB,
    const float* __restrict__ Y,    // EPI==2 multiplier, ld = ldOut
    int ldOut) {
  const int tid  = threadIdx.x;
  const int wave = tid >> 5;          // 0..7
  const int lane = tid & 31;
  const int l15  = lane & 15;
  const int hi   = lane >> 4;         // half-wave select
  const int wm   = wave >> 1;         // 0..3  (M direction)
  const int wn   = wave & 1;          // 0..1  (N direction)
  const int m0   = blockIdx.x * 128 + wm * 32;
  const int n0   = blockIdx.y * 128 + wn * 64;

  // A fragment K offset per half-wave (contiguous 16 K per half-wave, matches B)
  const int kbA = hi ? 16 : 0;
  // B fragment K offset per half-wave (column-per-lane layout)
  const int koB = hi ? 16 : 0;

  v8f acc[2][4];
#pragma unroll
  for (int i = 0; i < 2; ++i)
#pragma unroll
    for (int j = 0; j < 4; ++j)
#pragma unroll
      for (int r = 0; r < 8; ++r) acc[i][j][r] = 0.0f;

  for (int k0 = 0; k0 < K; k0 += 32) {
    v16bf afrag[2];
#pragma unroll
    for (int i = 0; i < 2; ++i) {
      const bf16_t* ap = A + (size_t)(m0 + i * 16 + l15) * K + (k0 + kbA);
      v8bf lo  = *(const v8bf*)(ap);
      v8bf hi8 = *(const v8bf*)(ap + 8);
      afrag[i] = cat8(lo, hi8);
    }
    v16bf bfrag[4];
#pragma unroll
    for (int j = 0; j < 4; ++j) {
      const bf16_t* bp = BT + (size_t)(n0 + j * 16 + l15) * K + (k0 + koB);
      bfrag[j] = *(const v16bf*)bp;
    }
#pragma unroll
    for (int i = 0; i < 2; ++i)
#pragma unroll
      for (int j = 0; j < 4; ++j)
        acc[i][j] = __builtin_amdgcn_wmma_f32_16x16x32_bf16(
            false, afrag[i], false, bfrag[j], (short)0, acc[i][j],
            false, false);
  }

  // Epilogue: D VGPR r -> row = base + hi*8 + r, col = base + (lane&15)
#pragma unroll
  for (int i = 0; i < 2; ++i) {
#pragma unroll
    for (int j = 0; j < 4; ++j) {
      const int colBase = n0 + j * 16 + l15;
#pragma unroll
      for (int r = 0; r < 8; ++r) {
        const int row = m0 + i * 16 + hi * 8 + r;
        const size_t idx = (size_t)row * ldOut + colBase;
        const float v = acc[i][j][r];
        if (EPI == 0) {
          outF[idx] = v;
        } else if (EPI == 1) {
          // silu(v) = v * sigmoid(v); v_exp + v_rcp + v_mul (bf16-accurate)
          const float s = v * __builtin_amdgcn_rcpf(1.0f + __expf(-v));
          outB[idx] = (bf16_t)s;
        } else {
          outB[idx] = (bf16_t)(v * Y[idx]);
        }
      }
    }
  }
}

// ---------------------------------------------------------------------------
// Sequential SSD scan. One block per (batch, head); 256 threads.
// Thread owns p = tid/4 (head-dim row) and 16 states n in [ (tid&3)*16, +16 ).
// B/C/X/dt staged through LDS in chunks of 16 timesteps.
// ---------------------------------------------------------------------------
__global__ __launch_bounds__(256) void ssd_scan_kernel(
    const float* __restrict__ ssm,      // (MTOT, LDSSM)
    const float* __restrict__ dt_bias,  // (NHEADS)
    const float* __restrict__ A_log,    // (NHEADS)
    const float* __restrict__ D_param,  // (NHEADS)
    float* __restrict__ y,              // (MTOT, DINNER)
    float* __restrict__ h_last) {       // (BATCH, NHEADS, 64, 64)
  const int bh = blockIdx.x;
  const int b  = bh >> 3;
  const int h  = bh & 7;
  const int tid = threadIdx.x;
  const int p  = tid >> 2;          // 0..63
  const int ng = tid & 3;
  const int nb = ng * 16;           // owned state base

  const float negA = -__expf(A_log[h]);
  const float dtb  = dt_bias[h];
  const float Dp   = D_param[h];

  float hreg[16];
#pragma unroll
  for (int j = 0; j < 16; ++j) hreg[j] = 0.0f;

  __shared__ float Xs[16][64];
  __shared__ float Bs[16][64];
  __shared__ float Cs[16][64];
  __shared__ float dts[16];

  const int r = tid >> 4;           // chunk row this thread helps load
  const int w = tid & 15;

  for (int t0 = 0; t0 < SEQ; t0 += 16) {
    const float* rowp = ssm + (size_t)(b * SEQ + t0 + r) * LDSSM;
#pragma unroll
    for (int q = 0; q < 4; ++q) {
      const int c = w * 4 + q;
      Bs[r][c] = rowp[c];
      Cs[r][c] = rowp[DSTATE + c];
      Xs[r][c] = rowp[2 * DSTATE + NHEADS + h * HEADDIM + c];
    }
    if (w == 0) dts[r] = rowp[2 * DSTATE + h];
    __syncthreads();

    for (int s = 0; s < 16; ++s) {
      const float draw  = dts[s] + dtb;
      const float delta = (draw > 20.0f) ? draw : log1pf(__expf(draw));
      const float dA    = __expf(delta * negA);
      const float xv    = Xs[s][p];
      float ysum = 0.0f;
#pragma unroll
      for (int j = 0; j < 16; ++j) {
        const float dB = delta * Bs[s][nb + j];
        hreg[j] = dA * hreg[j] + xv * dB;
        ysum += hreg[j] * Cs[s][nb + j];
      }
      // reduce the 4 partial sums that share p (consecutive lanes, wave32)
      ysum += __shfl_xor(ysum, 1, 32);
      ysum += __shfl_xor(ysum, 2, 32);
      if (ng == 0)
        y[(size_t)(b * SEQ + t0 + s) * DINNER + h * HEADDIM + p] =
            ysum + Dp * xv;
    }
    __syncthreads();
  }

  const size_t base = (((size_t)(b * NHEADS + h) * HEADDIM) + p) * DSTATE + nb;
#pragma unroll
  for (int j = 0; j < 16; ++j) h_last[base + j] = hreg[j];
}

// ---------------------------------------------------------------------------
// Host-side launch
// ---------------------------------------------------------------------------
extern "C" void kernel_launch(void* const* d_in, const int* in_sizes, int n_in,
                              void* d_out, int out_size, void* d_ws,
                              size_t ws_size, hipStream_t stream) {
  const float* x       = (const float*)d_in[0];  // (2,4096,1024)
  const float* Wz      = (const float*)d_in[1];  // (1024,2048)
  const float* Win     = (const float*)d_in[2];  // (1024,648)
  const float* Wzg     = (const float*)d_in[3];  // (2048,512)
  const float* Wout    = (const float*)d_in[4];  // (512,1024)
  const float* dt_bias = (const float*)d_in[5];  // (8)
  const float* A_log   = (const float*)d_in[6];  // (8)
  const float* D_param = (const float*)d_in[7];  // (8)

  float* out    = (float*)d_out;                       // (2,4096,1024)
  float* h_last = out + (size_t)MTOT * DMODEL;         // (2,8,64,64)

  // workspace carve-out (256B aligned)
  char* ws = (char*)d_ws;
  size_t off = 0;
  auto alloc = [&](size_t bytes) -> char* {
    char* p = ws + off;
    off = (off + bytes + 255) & ~(size_t)255;
    return p;
  };
  bf16_t* xb    = (bf16_t*)alloc((size_t)MTOT * DMODEL * 2);
  bf16_t* WzT   = (bf16_t*)alloc((size_t)DZ * DMODEL * 2);
  bf16_t* WinT  = (bf16_t*)alloc((size_t)LDSSM * DMODEL * 2);
  bf16_t* WzgT  = (bf16_t*)alloc((size_t)DINNER * DZ * 2);
  bf16_t* WoutT = (bf16_t*)alloc((size_t)DMODEL * DINNER * 2);
  bf16_t* zs    = (bf16_t*)alloc((size_t)MTOT * DZ * 2);      // silu(x@Wz)
  float*  ssm   = (float*) alloc((size_t)MTOT * LDSSM * 4);   // x@Win (padded)
  float*  yb    = (float*) alloc((size_t)MTOT * DINNER * 4);  // scan output
  bf16_t* gb    = (bf16_t*)alloc((size_t)MTOT * DINNER * 2);  // y * z_gated

  // 1. conversions
  cvt_bf16_kernel<<<4096, 256, 0, stream>>>(x, xb, (size_t)MTOT * DMODEL);
  transpose_cvt_kernel<<<2048, 256, 0, stream>>>(Wz, WzT, DMODEL, DZ, DZ);
  transpose_cvt_kernel<<<768, 256, 0, stream>>>(Win, WinT, DMODEL, NSSM, LDSSM);
  transpose_cvt_kernel<<<1024, 256, 0, stream>>>(Wzg, WzgT, DZ, DINNER, DINNER);
  transpose_cvt_kernel<<<512, 256, 0, stream>>>(Wout, WoutT, DINNER, DMODEL,
                                                DMODEL);

  // 2. zs = silu(x @ Wz)        [8192 x 1024 x 2048]
  gemm_bf16_kernel<1><<<dim3(MTOT / 128, DZ / 128), 256, 0, stream>>>(
      xb, WzT, DMODEL, nullptr, zs, nullptr, DZ);

  // 3. ssm = x @ Win (padded)   [8192 x 1024 x 768]
  gemm_bf16_kernel<0><<<dim3(MTOT / 128, LDSSM / 128), 256, 0, stream>>>(
      xb, WinT, DMODEL, ssm, nullptr, nullptr, LDSSM);

  // 4. sequential SSD scan -> y, h_last
  ssd_scan_kernel<<<BATCH * NHEADS, 256, 0, stream>>>(ssm, dt_bias, A_log,
                                                      D_param, yb, h_last);

  // 5. gb = bf16( (zs @ Wzg) * y )   [8192 x 2048 x 512]
  gemm_bf16_kernel<2><<<dim3(MTOT / 128, DINNER / 128), 256, 0, stream>>>(
      zs, WzgT, DZ, nullptr, gb, yb, DINNER);

  // 6. out = gb @ Wout          [8192 x 512 x 1024]
  gemm_bf16_kernel<0><<<dim3(MTOT / 128, DMODEL / 128), 256, 0, stream>>>(
      gb, WoutT, DINNER, out, nullptr, nullptr, DMODEL);
}
